// KmeansVectorQuantizer_76046690943037
// MI455X (gfx1250) — matrix operand order repaired and checked by reading, not compiled, not run on hardware
//
#include <hip/hip_runtime.h>
#include <hip/hip_bf16.h>
#include <stdint.h>

// ---------------- problem constants ----------------
#define GQ 8
#define DQ 64
#define VQ 1024
#define BQ 8
#define TQ 2048
#define BT (BQ * TQ)   // 16384 tokens

// ---------------- types ----------------
typedef __attribute__((ext_vector_type(16))) __bf16 v16bf;
typedef __attribute__((ext_vector_type(8)))  float  v8f;
typedef __attribute__((ext_vector_type(4)))  unsigned int u32x4;
typedef __attribute__((ext_vector_type(4)))  unsigned int v4u;
typedef __attribute__((ext_vector_type(8)))  int v8i;
typedef __attribute__((ext_vector_type(4)))  int v4i;

union BF16Frag {
    v16bf v;
    unsigned short s[16];
    u32x4 q[2];
};

// round-to-nearest-even fp32 -> bf16 (bit trick, no __bf16 arithmetic needed)
__device__ __host__ inline unsigned short f2bf(float f) {
    union { float f; unsigned u; } c; c.f = f;
    unsigned r = c.u + 0x7FFFu + ((c.u >> 16) & 1u);
    return (unsigned short)(r >> 16);
}
__device__ __host__ inline float bf2f(unsigned short h) {
    union { unsigned u; float f; } c; c.u = ((unsigned)h) << 16;
    return c.f;
}

__device__ inline v8f wmma_bf16(v16bf a, v16bf b, v8f c) {
    // (neg_a, A, neg_b, B, c_mod, C, reuse_a, reuse_b)
    return __builtin_amdgcn_wmma_f32_16x16x32_bf16(false, a, false, b, (short)0, c,
                                                   false, false);
}

#if defined(__gfx1250__)
// 1-D TDM load: nelems bf16 elements, contiguous, global -> LDS.
// Descriptor layout per CDNA5 ISA.
// This toolchain's builtin takes 6 args:
//   (uint32x4 g0, int32x8 g1, int32x4 g2, int32x4 g3, int32x8 extra, i32 cpol)
__device__ inline void tdm_load_1d(unsigned lds_off, const void* gptr, int nelems) {
    unsigned long long ga = (unsigned long long)(uintptr_t)gptr;
    v4u g0;
    g0.x = 1u;                                        // count=1, is_restore=0, no gather
    g0.y = lds_off;                                   // lds_addr [63:32]
    g0.z = (unsigned)(ga & 0xFFFFFFFFu);              // global_addr [95:64]
    g0.w = ((unsigned)((ga >> 32) & 0x01FFFFFFu))     // global_addr [120:96]
           | 0x80000000u;                             // type=2 ("image") [127:126]
    v8i g1;
    g1[0] = (int)(1u << 16);                          // wg_mask=0, data_size=1 (2B)
    g1[1] = (int)(((unsigned)nelems & 0xFFFFu) << 16);        // tensor_dim0 lo16
    g1[2] = (int)((((unsigned)nelems >> 16) & 0xFFFFu)        // tensor_dim0 hi16
                  | (1u << 16));                              // tensor_dim1 = 1
    g1[3] = (int)(((unsigned)nelems & 0xFFFFu) << 16);        // tile_dim0 = nelems
    g1[4] = 0;                                        // tile_dim1=0, tile_dim2=0
    g1[5] = nelems;                                   // tensor_dim0_stride lo32
    g1[6] = 0;
    g1[7] = 0;
    v4i g2 = {0, 0, 0, 0};
    v4i g3 = {0, 0, 0, 0};
    v8i gx = {0, 0, 0, 0, 0, 0, 0, 0};
    __builtin_amdgcn_tensor_load_to_lds(g0, g1, g2, g3, gx, 0);
}
#endif

// ---------------- kernel 0: zero accumulators ----------------
__global__ void vq_init(float* accs) {
    accs[0] = 0.0f;   // loss numerator
    accs[1] = 0.0f;   // denom
}

// ---------------- kernel 1: denom = sum(1 - padding) ----------------
__global__ __launch_bounds__(256) void vq_denom(const int* __restrict__ padd,
                                                float* __restrict__ denom_acc) {
    int i = blockIdx.x * blockDim.x + threadIdx.x;
    float v = (i < BT) ? (1.0f - (float)padd[i]) : 0.0f;
    for (int off = 16; off; off >>= 1) v += __shfl_xor(v, off, 32);
    if ((threadIdx.x & 31) == 0) atomicAdd(denom_acc, v);
}

// ---------------- kernel 2: codebook -> bf16 hi/lo + ||c||^2 ----------------
// cbhl layout: [G][V][128] ushort, row = 64 hi then 64 lo.
__global__ __launch_bounds__(256) void vq_prep_cb(const float* __restrict__ cb,
                                                  unsigned short* __restrict__ cbhl,
                                                  float* __restrict__ csq) {
    int wid = (blockIdx.x * blockDim.x + threadIdx.x) >> 5;  // one wave per (v,g)
    int lane = threadIdx.x & 31;
    if (wid >= VQ * GQ) return;
    int v = wid / GQ, g = wid % GQ;
    unsigned short* row = cbhl + ((size_t)g * VQ + v) * 128;
    float ss = 0.0f;
    for (int j = 0; j < 2; ++j) {
        int d = lane * 2 + j;
        float f = cb[((size_t)v * GQ + g) * DQ + d];
        unsigned short h = f2bf(f);
        row[d]      = h;
        row[64 + d] = f2bf(f - bf2f(h));
        ss += f * f;
    }
    for (int off = 16; off; off >>= 1) ss += __shfl_xor(ss, off, 32);
    if (lane == 0) csq[g * VQ + v] = ss;
}

// ---------------- kernel 3: WMMA distance + argmin ----------------
// grid: (BT/128, G); block: 256 (8 waves); wave w handles token tile blockIdx.x*8+w.
__global__ __launch_bounds__(256) void vq_main(const float* __restrict__ x,
                                               const int* __restrict__ padd,
                                               const unsigned short* __restrict__ cbhl,
                                               const float* __restrict__ csq,
                                               int* __restrict__ ids_ws,
                                               float* __restrict__ out_ids) {
    __shared__ __align__(16) unsigned short sbuf[2][16 * 128];  // double buffer, 8KB

    const int g    = blockIdx.y;
    const int wave = threadIdx.x >> 5;
    const int lane = threadIdx.x & 31;
    const int half = lane >> 4;
    const int mrow = lane & 15;
    const int tile = blockIdx.x * 8 + wave;
    const int token0 = tile * 16;

    // ---- A fragments: 16 tokens x K=64, bf16 hi/lo ----
    // A k order chosen here (B matches): lane half selects contiguous K {16h..16h+15}.
    BF16Frag a_hi[2], a_lo[2];
    {
        const float* xp = x + ((size_t)(token0 + mrow) * GQ + g) * DQ;
        for (int c = 0; c < 2; ++c)
            for (int run = 0; run < 2; ++run)
                for (int i = 0; i < 8; ++i) {
                    int k = c * 32 + half * 16 + run * 8 + i;
                    float f = xp[k];
                    unsigned short h = f2bf(f);
                    a_hi[c].s[run * 8 + i] = h;
                    a_lo[c].s[run * 8 + i] = f2bf(f - bf2f(h));
                }
    }

    float bestv[8];
    int   besti[8];
    for (int r = 0; r < 8; ++r) { bestv[r] = -3.4e38f; besti[r] = 0; }

#if defined(__gfx1250__)
    if (threadIdx.x < 32)
        tdm_load_1d((unsigned)(uintptr_t)(void*)&sbuf[0][0],
                    cbhl + (size_t)g * VQ * 128, 16 * 128);
#endif

    for (int ch = 0; ch < 64; ++ch) {
        const int pb = ch & 1;
#if defined(__gfx1250__)
        if (threadIdx.x < 32) __builtin_amdgcn_s_wait_tensorcnt(0);
#endif
        __syncthreads();   // buffer pb is ready; buffer pb^1 free to overwrite
#if defined(__gfx1250__)
        if (ch + 1 < 64 && threadIdx.x < 32)
            tdm_load_1d((unsigned)(uintptr_t)(void*)&sbuf[pb ^ 1][0],
                        cbhl + ((size_t)g * VQ + (ch + 1) * 16) * 128, 16 * 128);
#else
        if (true) {  // portable fallback staging (not the gfx1250 path)
            for (int e = threadIdx.x; e < 16 * 128; e += 256)
                sbuf[pb][e] = cbhl[((size_t)g * VQ + ch * 16) * 128 + e];
            __syncthreads();
        }
#endif
        // ---- B fragments from LDS: column = mrow, K = kc*32 + half*16 + e ----
        BF16Frag b_hi[2], b_lo[2];
        for (int kc = 0; kc < 2; ++kc) {
            const unsigned base = (unsigned)mrow * 128 + kc * 32 + half * 16;
            const u32x4* ph = (const u32x4*)&sbuf[pb][base];
            const u32x4* pl = (const u32x4*)&sbuf[pb][base + 64];
            b_hi[kc].q[0] = ph[0]; b_hi[kc].q[1] = ph[1];
            b_lo[kc].q[0] = pl[0]; b_lo[kc].q[1] = pl[1];
        }

        v8f acc = {0.f, 0.f, 0.f, 0.f, 0.f, 0.f, 0.f, 0.f};
        acc = wmma_bf16(a_hi[0].v, b_hi[0].v, acc);
        acc = wmma_bf16(a_hi[1].v, b_hi[1].v, acc);
        acc = wmma_bf16(a_hi[0].v, b_lo[0].v, acc);
        acc = wmma_bf16(a_hi[1].v, b_lo[1].v, acc);
        acc = wmma_bf16(a_lo[0].v, b_hi[0].v, acc);
        acc = wmma_bf16(a_lo[1].v, b_hi[1].v, acc);

        const int n = ch * 16 + mrow;                 // code column this lane holds
        const float bias = -0.5f * csq[g * VQ + n];   // argmax(dot - ||c||^2/2)
        for (int r = 0; r < 8; ++r) {
            float s = acc[r] + bias;
            if (s > bestv[r]) { bestv[r] = s; besti[r] = n; }
        }
    }

    // argmax reduce across the 16 lanes holding the same M rows; prefer lower id
    for (int off = 8; off; off >>= 1)
        for (int r = 0; r < 8; ++r) {
            float ov = __shfl_xor(bestv[r], off, 32);
            int   oi = __shfl_xor(besti[r], off, 32);
            if (ov > bestv[r] || (ov == bestv[r] && oi < besti[r])) {
                bestv[r] = ov; besti[r] = oi;
            }
        }

    if (mrow == 0) {
        for (int r = 0; r < 8; ++r) {
            int m = r + half * 8;                 // C layout: lanes16-31 hold M=r+8
            int token = token0 + m;
            int id = besti[r];
            ids_ws[token * GQ + g] = id;
            int pad = padd[token];
            out_ids[(size_t)token * GQ + g] = pad ? -1.0f : (float)id;
        }
    }
}

// ---------------- kernel 4: gather fp32 code, write quantized*mask, loss ----
__global__ __launch_bounds__(256) void vq_quant_loss(const float* __restrict__ x,
                                                     const int* __restrict__ padd,
                                                     const float* __restrict__ cb,
                                                     const int* __restrict__ ids_ws,
                                                     float* __restrict__ out_q,
                                                     float* __restrict__ loss_acc) {
    int wid  = (blockIdx.x * blockDim.x + threadIdx.x) >> 5;  // one wave per (token,g)
    int lane = threadIdx.x & 31;
    if (wid >= BT * GQ) return;
    int token = wid / GQ, g = wid % GQ;
    int pad = padd[token];
    int id  = ids_ws[wid];
    float m = pad ? 0.0f : 1.0f;
    const float* xp = x   + (size_t)wid * DQ;
    const float* cp = cb  + ((size_t)id * GQ + g) * DQ;
    float*       qp = out_q + (size_t)wid * DQ;
    float ls = 0.0f;
    for (int j = 0; j < 2; ++j) {
        int d = lane * 2 + j;
        float c = cp[d], xv = xp[d];
        qp[d] = c * m;
        float e = xv - c;
        ls += e * e;
    }
    for (int off = 16; off; off >>= 1) ls += __shfl_xor(ls, off, 32);
    if (lane == 0 && !pad) atomicAdd(loss_acc, ls);
}

// ---------------- kernel 5: finalize losses ----------------
__global__ void vq_final(const float* __restrict__ accs, float* __restrict__ out_sc) {
    float den = accs[1];
    float l = (den > 0.0f) ? (accs[0] / den) : 0.0f;
    out_sc[0] = l;          // kmeans_loss
    out_sc[1] = l;          // commitment_loss (numerically identical)
    out_sc[2] = 2.0f * l;   // total (BETA = 1.0)
}

// ---------------- host launcher ----------------
extern "C" void kernel_launch(void* const* d_in, const int* in_sizes, int n_in,
                              void* d_out, int out_size, void* d_ws, size_t ws_size,
                              hipStream_t stream) {
    (void)in_sizes; (void)n_in; (void)out_size; (void)ws_size;
    const float* x    = (const float*)d_in[0];  // (8,2048,512) fp32
    const int*   padd = (const int*)d_in[1];    // (8,2048) int32
    const float* cb   = (const float*)d_in[2];  // (1024,8,64) fp32

    float* out      = (float*)d_out;
    float* out_ids  = out;                       // BT*G
    float* out_q    = out + (size_t)BT * GQ;     // BT*G*D
    float* out_sc   = out + (size_t)BT * GQ + (size_t)BT * GQ * DQ;  // 3 scalars

    char* ws = (char*)d_ws;
    unsigned short* cbhl = (unsigned short*)ws;                       // 2 MB
    float* csq   = (float*)(ws + (size_t)GQ * VQ * 128 * 2);          // 32 KB
    int*   idsws = (int*)((char*)csq + (size_t)GQ * VQ * 4);          // 512 KB
    float* accs  = (float*)((char*)idsws + (size_t)BT * GQ * 4);      // 2 floats

    vq_init<<<1, 1, 0, stream>>>(accs);
    vq_denom<<<(BT + 255) / 256, 256, 0, stream>>>(padd, accs + 1);
    vq_prep_cb<<<(VQ * GQ * 32) / 256, 256, 0, stream>>>(cb, cbhl, csq);

    dim3 grid(BT / (16 * 8), GQ);   // (128, 8)
    vq_main<<<grid, 256, 0, stream>>>(x, padd, cbhl, csq, idsws, out_ids);

    vq_quant_loss<<<(BT * GQ * 32) / 256, 256, 0, stream>>>(x, padd, cb, idsws,
                                                            out_q, accs);
    vq_final<<<1, 1, 0, stream>>>(accs, out_sc);
}
